// AttentionalSplatting_12481174962698
// MI455X (gfx1250) — matrix operand order repaired and checked by reading, not compiled, not run on hardware
//
#include <hip/hip_runtime.h>
#include <hip/hip_bf16.h>

// Problem constants (from the reference)
#define TT   16
#define HWN  1024
#define MM   512
#define DM   1024
#define DK   512
#define NH   8
#define HD   64     // q/k head dim
#define VD   128    // v head dim

typedef __attribute__((ext_vector_type(16))) __bf16          v16bf;
typedef __attribute__((ext_vector_type(16))) unsigned short  v16u;
typedef __attribute__((ext_vector_type(8)))  float           v8f;

__device__ __forceinline__ unsigned short f2bf(float f) {
  unsigned int u = __float_as_uint(f);
  u += 0x7FFFu + ((u >> 16) & 1u);          // round-to-nearest-even
  return (unsigned short)(u >> 16);
}

__device__ __forceinline__ v8f wmma_bf16(v16u a, v16u b, v8f c) {
  return __builtin_amdgcn_wmma_f32_16x16x32_bf16(
      false, __builtin_bit_cast(v16bf, a),
      false, __builtin_bit_cast(v16bf, b),
      (short)0, c, false, false);
}

// LDS byte-offset of a __shared__ pointer (addrspace(3) pointers are 32-bit
// offsets into the workgroup's LDS allocation on AMDGPU).
__device__ __forceinline__ unsigned to_lds_off(void* p) {
  return (unsigned)(unsigned long long)(__attribute__((address_space(3))) char*)p;
}

// Async memory -> LDS copy, 16 bytes per lane, tracked by ASYNCcnt.
__device__ __forceinline__ void async_g2l_b128(unsigned lds_off, const void* gptr) {
  asm volatile("global_load_async_to_lds_b128 %0, %1, off"
               :: "v"(lds_off), "v"(gptr) : "memory");
}
__device__ __forceinline__ void wait_asynccnt0() {
  asm volatile("s_wait_asynccnt 0" ::: "memory");
}

// ---------------------------------------------------------------------------
// Generic bf16 WMMA GEMM: C[M,N] = A[M,K] * B[K,N], fp32 accumulate.
// A is fp32 (converted on load) or bf16 (ABF16=true); B is fp32 (converted).
// Block tile 128x64, 8 waves, each wave 32x32 (2x2 WMMA tiles), K-step 32.
// ---------------------------------------------------------------------------
template <bool ABF16>
__global__ __launch_bounds__(256) void gemm_wmma(
    const void* __restrict__ Aany, const float* __restrict__ B,
    float* __restrict__ C, int Mdim, int Ndim, int Kdim) {
  __shared__ unsigned short sA [128][40];   // [m][k]  (row-major, padded)
  __shared__ unsigned short sBT[ 64][40];   // [n][k]  (transposed, padded)

  const float*          Af = (const float*)Aany;
  const unsigned short* Ah = (const unsigned short*)Aany;

  const int tid  = threadIdx.x;
  const int lane = tid & 31;
  const int wave = tid >> 5;
  const int wm   = wave & 3;          // 4 row groups of 32
  const int wn   = wave >> 2;         // 2 col groups of 32
  const long bM  = (long)blockIdx.y * 128;
  const long bN  = (long)blockIdx.x * 64;
  const int m    = lane & 15;
  const int half = lane >> 4;

  v8f acc[2][2];
#pragma unroll
  for (int tr = 0; tr < 2; ++tr)
#pragma unroll
    for (int tc = 0; tc < 2; ++tc) acc[tr][tc] = (v8f)0.0f;

  const int arowL = tid >> 1, acb = (tid & 1) << 4;   // A coop load: 16 elems
  const int bkr   = tid >> 3, bcb = (tid & 7) << 3;   // B coop load: 8 elems

  for (int k0 = 0; k0 < Kdim; k0 += 32) {
    const long gA = (bM + arowL) * (long)Kdim + k0 + acb;
    const long gB = (long)(k0 + bkr) * Ndim + bN + bcb;
    {
#pragma unroll
      for (int j = 0; j < 16; ++j)
        sA[arowL][acb + j] = ABF16 ? Ah[gA + j] : f2bf(Af[gA + j]);
#pragma unroll
      for (int j = 0; j < 8; ++j) sBT[bcb + j][bkr] = f2bf(B[gB + j]);
    }
    // Start pulling the next K-tile toward the caches while we do WMMAs.
    if (k0 + 32 < Kdim) {
      if (ABF16) __builtin_prefetch((const void*)(Ah + gA + 32), 0, 0);
      else       __builtin_prefetch((const void*)(Af + gA + 32), 0, 0);
      __builtin_prefetch((const void*)(B + gB + (long)32 * Ndim), 0, 0);
    }
    __syncthreads();

    v16u au[2], bu[2];
    const int b8  = half * 8;
    const int b16 = half * 16;
#pragma unroll
    for (int tr = 0; tr < 2; ++tr) {
      const int ar = wm * 32 + tr * 16 + m;
#pragma unroll
      for (int e = 0; e < 8; ++e)  au[tr][e] = sA[ar][b8 + e];        // K=b8+e
#pragma unroll
      for (int e = 8; e < 16; ++e) au[tr][e] = sA[ar][b8 + e + 8];    // K=b8+8+e
    }
#pragma unroll
    for (int tc = 0; tc < 2; ++tc) {
      const int bc = wn * 32 + tc * 16 + m;
#pragma unroll
      for (int e = 0; e < 8; ++e)  bu[tc][e] = sBT[bc][b8 + e];       // K=b8+e
#pragma unroll
      for (int e = 8; e < 16; ++e) bu[tc][e] = sBT[bc][b8 + e + 8];   // K=b8+8+e
    }
#pragma unroll
    for (int tr = 0; tr < 2; ++tr)
#pragma unroll
      for (int tc = 0; tc < 2; ++tc)
        acc[tr][tc] = wmma_bf16(au[tr], bu[tc], acc[tr][tc]);
    __syncthreads();
  }

#pragma unroll
  for (int tr = 0; tr < 2; ++tr)
#pragma unroll
    for (int tc = 0; tc < 2; ++tc)
#pragma unroll
      for (int r = 0; r < 8; ++r) {
        const long grow = bM + wm * 32 + tr * 16 + half * 8 + r;
        const long gcol = bN + wn * 32 + tc * 16 + m;
        C[grow * Ndim + gcol] = acc[tr][tc][r];
      }
}

// ---------------------------------------------------------------------------
// Block reduction helper (wave32)
// ---------------------------------------------------------------------------
__device__ __forceinline__ float block_reduce_sum(float v, float* sbuf) {
#pragma unroll
  for (int msk = 16; msk >= 1; msk >>= 1) v += __shfl_xor(v, msk, 32);
  const int tid = threadIdx.x;
  if ((tid & 31) == 0) sbuf[tid >> 5] = v;
  __syncthreads();
  const int nw = (blockDim.x + 31) >> 5;
  if (tid < 32) {
    float x = (tid < nw) ? sbuf[tid] : 0.0f;
#pragma unroll
    for (int msk = 16; msk >= 1; msk >>= 1) x += __shfl_xor(x, msk, 32);
    if (tid == 0) sbuf[0] = x;
  }
  __syncthreads();
  const float r = sbuf[0];
  __syncthreads();
  return r;
}

// LayerNorm(Q) -> bf16.  One block per row, 256 threads, 2 elems each.
__global__ __launch_bounds__(256) void ln_q_kernel(
    const float* __restrict__ X, const float* __restrict__ gamma,
    unsigned short* __restrict__ Y) {
  __shared__ float sred[16];
  const long row = blockIdx.x;
  const int  d0  = threadIdx.x, d1 = threadIdx.x + 256;
  const float x0 = X[row * DK + d0], x1 = X[row * DK + d1];
  const float mu  = block_reduce_sum(x0 + x1, sred) * (1.0f / 512.0f);
  const float var = block_reduce_sum((x0 - mu) * (x0 - mu) + (x1 - mu) * (x1 - mu),
                                     sred) * (1.0f / 512.0f);
  const float inv = rsqrtf(var + 1e-6f);
  Y[row * DK + d0] = f2bf((x0 - mu) * inv * gamma[d0]);
  Y[row * DK + d1] = f2bf((x1 - mu) * inv * gamma[d1]);
}

// RoPE-2D + LayerNorm on K (D=512) -> bf16.  One block per (t,m) row,
// 256 threads = 256 rotation pairs.
__global__ __launch_bounds__(256) void rope_ln_k_kernel(
    const float* __restrict__ Kf, const float* __restrict__ tracks,
    const float* __restrict__ gamma, unsigned short* __restrict__ Kout) {
  __shared__ float sred[16];
  const long row = blockIdx.x;                       // t*MM + m
  const int  p   = threadIdx.x;                      // pair id
  const float px = tracks[row * 2], py = tracks[row * 2 + 1];
  const int   j  = p & 127;                          // quarter = 128
  const float theta = __powf(10000.0f, -(float)j * (1.0f / 128.0f));
  const float f  = (p < 128 ? px : py) * theta;
  float s, c; __sincosf(f, &s, &c);
  const int d1 = (p < 128) ? (2 * p) : (256 + 2 * (p - 128));
  const float x1 = Kf[row * DK + d1], x2 = Kf[row * DK + d1 + 1];
  const float y1 = x1 * c - x2 * s, y2 = x1 * s + x2 * c;
  const float mu  = block_reduce_sum(y1 + y2, sred) * (1.0f / 512.0f);
  const float var = block_reduce_sum((y1 - mu) * (y1 - mu) + (y2 - mu) * (y2 - mu),
                                     sred) * (1.0f / 512.0f);
  const float inv = rsqrtf(var + 1e-6f);
  Kout[row * DK + d1]     = f2bf((y1 - mu) * inv * gamma[d1]);
  Kout[row * DK + d1 + 1] = f2bf((y2 - mu) * inv * gamma[d1 + 1]);
}

// RoPE-2D on V (D=1024) -> bf16.  One block per (t,m) row, 512 pairs.
__global__ __launch_bounds__(512) void rope_v_kernel(
    const float* __restrict__ V, const float* __restrict__ tracks,
    unsigned short* __restrict__ Vout) {
  const long row = blockIdx.x;
  const int  p   = threadIdx.x;
  const float px = tracks[row * 2], py = tracks[row * 2 + 1];
  const int   j  = p & 255;                          // quarter = 256
  const float theta = __powf(10000.0f, -(float)j * (1.0f / 256.0f));
  const float f  = (p < 256 ? px : py) * theta;
  float s, c; __sincosf(f, &s, &c);
  const int d1 = (p < 256) ? (2 * p) : (512 + 2 * (p - 256));
  const float x1 = V[row * DM + d1], x2 = V[row * DM + d1 + 1];
  Vout[row * DM + d1]     = f2bf(x1 * c - x2 * s);
  Vout[row * DM + d1 + 1] = f2bf(x1 * s + x2 * c);
}

// Tiled transpose Vbf [t][key][d] -> VbfT [t][d][key] (both bf16), coalesced
// on load and store, so the attention kernel can use async B128 copies.
__global__ __launch_bounds__(256) void vtrans_kernel(
    const unsigned short* __restrict__ Vbf, unsigned short* __restrict__ VbfT) {
  __shared__ unsigned short sT[64][72];
  const int  t   = blockIdx.z;
  const long d0  = (long)blockIdx.y * 64;   // over DM
  const long kk0 = (long)blockIdx.x * 64;   // over MM (keys)
  const int  tid = threadIdx.x;
  const int  r   = tid >> 2;                // 0..63
  const int  cb  = (tid & 3) << 4;          // 0,16,32,48
#pragma unroll
  for (int j = 0; j < 16; ++j)
    sT[r][cb + j] = Vbf[((long)t * MM + kk0 + r) * DM + d0 + cb + j];
  __syncthreads();
#pragma unroll
  for (int j = 0; j < 16; ++j)
    VbfT[((long)t * DM + d0 + r) * MM + kk0 + cb + j] = sT[cb + j][r];
}

// ---------------------------------------------------------------------------
// Attention: per block (t, h, 64 q-rows); 4 waves x 16 q-rows.
// Two-pass softmax with Gaussian-splat bias; WMMA for QK^T and P*V.
// V chunks are staged into LDS with async global->LDS B128 copies.
// ---------------------------------------------------------------------------
__device__ __forceinline__ v8f score_tile(const unsigned short* __restrict__ Kbase,
                                          int key, const v16u* aq, int half) {
  const unsigned short* krow = Kbase + (long)key * DK;
  const int b8 = half * 8;
  v16u b0, b1;
#pragma unroll
  for (int e = 0; e < 16; ++e) {
    const int kk = (e < 8) ? (b8 + e) : (b8 + e + 8);
    b0[e] = krow[kk]; b1[e] = krow[32 + kk];
  }
  v8f c = (v8f)0.0f;
  c = wmma_bf16(aq[0], b0, c);
  c = wmma_bf16(aq[1], b1, c);
  return c;
}

__global__ __launch_bounds__(128) void attn_kernel(
    const unsigned short* __restrict__ Qbf, const unsigned short* __restrict__ Kbf,
    const unsigned short* __restrict__ VbfT, const float* __restrict__ tracks,
    const float* __restrict__ fpos, unsigned short* __restrict__ Sout) {
  __shared__ unsigned short sP [4][16][40];            // per-wave P [16 q x 32 keys]
  alignas(16) __shared__ unsigned short sVT[128][40];  // V^T chunk [col][key]

  const int t = blockIdx.z, h = blockIdx.y;
  const int tid  = threadIdx.x;
  const int lane = tid & 31;
  const int wave = tid >> 5;
  const int q0   = blockIdx.x * 64 + wave * 16;
  const int m    = lane & 15;
  const int half = lane >> 4;
  const int b8   = half * 8;

  const unsigned short* Kbase = Kbf + (long)(t * MM) * DK + h * HD;
  const float* trk = tracks + (long)(t * MM) * 2;

  // Q A-fragments (head dims 0..63 = two K-steps of 32), persistent.
  v16u aq[2];
  {
    const unsigned short* qrow = Qbf + ((long)(t * HWN) + q0 + m) * DK + h * HD;
#pragma unroll
    for (int ks = 0; ks < 2; ++ks) {
#pragma unroll
      for (int e = 0; e < 8; ++e)  aq[ks][e] = qrow[ks * 32 + b8 + e];
#pragma unroll
      for (int e = 8; e < 16; ++e) aq[ks][e] = qrow[ks * 32 + b8 + e + 8];
    }
  }
  // Feature positions for the 8 C-rows this lane owns.
  float fpx[8], fpy[8];
#pragma unroll
  for (int r = 0; r < 8; ++r) {
    const int q = q0 + half * 8 + r;
    fpx[r] = fpos[q * 2]; fpy[r] = fpos[q * 2 + 1];
  }

  // ---- pass 1: global row max of (0.125*QK^T - 2*d2) ----
  float mrow[8];
#pragma unroll
  for (int r = 0; r < 8; ++r) mrow[r] = -1e30f;
  for (int nt = 0; nt < 32; ++nt) {
    const int key = nt * 16 + m;
    const v8f c = score_tile(Kbase, key, aq, half);
    const float tx = trk[key * 2], ty = trk[key * 2 + 1];
#pragma unroll
    for (int r = 0; r < 8; ++r) {
      const float dx = fpx[r] - tx, dy = fpy[r] - ty;
      const float s = c[r] * 0.125f - 2.0f * (dx * dx + dy * dy);
      mrow[r] = fmaxf(mrow[r], s);
    }
  }
#pragma unroll
  for (int r = 0; r < 8; ++r)
#pragma unroll
    for (int msk = 8; msk >= 1; msk >>= 1)
      mrow[r] = fmaxf(mrow[r], __shfl_xor(mrow[r], msk, 32));

  // ---- pass 2: per-32-key chunk, exp, P->LDS, async V stage, WMMA P*V ----
  float lrow[8];
#pragma unroll
  for (int r = 0; r < 8; ++r) lrow[r] = 0.0f;
  v8f o[8];
#pragma unroll
  for (int tc = 0; tc < 8; ++tc) o[tc] = (v8f)0.0f;

  for (int kc = 0; kc < 16; ++kc) {
    // Kick off the async V^T chunk copy first: 128 cols x 64B, one col/thread.
    {
      const unsigned short* src =
          VbfT + ((long)(t * DM + h * VD + tid)) * MM + kc * 32;
      const unsigned dst = to_lds_off((void*)&sVT[tid][0]);
#pragma unroll
      for (int j = 0; j < 4; ++j)
        async_g2l_b128(dst + j * 16, (const void*)(src + j * 8));
    }
#pragma unroll
    for (int ntl = 0; ntl < 2; ++ntl) {
      const int key = (kc * 2 + ntl) * 16 + m;
      const v8f c = score_tile(Kbase, key, aq, half);
      const float tx = trk[key * 2], ty = trk[key * 2 + 1];
#pragma unroll
      for (int r = 0; r < 8; ++r) {
        const float dx = fpx[r] - tx, dy = fpy[r] - ty;
        const float s = c[r] * 0.125f - 2.0f * (dx * dx + dy * dy);
        const float p = __expf(s - mrow[r]);
        lrow[r] += p;
        sP[wave][half * 8 + r][ntl * 16 + m] = f2bf(p);
      }
    }
    wait_asynccnt0();
    __syncthreads();

    v16u ap;
#pragma unroll
    for (int e = 0; e < 8; ++e)  ap[e] = sP[wave][m][b8 + e];
#pragma unroll
    for (int e = 8; e < 16; ++e) ap[e] = sP[wave][m][b8 + e + 8];
    const int b16 = half * 16;
#pragma unroll
    for (int tc = 0; tc < 8; ++tc) {
      v16u bv;
#pragma unroll
      for (int e = 0; e < 8; ++e)  bv[e] = sVT[tc * 16 + m][b8 + e];
#pragma unroll
      for (int e = 8; e < 16; ++e) bv[e] = sVT[tc * 16 + m][b8 + e + 8];
      o[tc] = wmma_bf16(ap, bv, o[tc]);
    }
    __syncthreads();
  }

#pragma unroll
  for (int r = 0; r < 8; ++r)
#pragma unroll
    for (int msk = 8; msk >= 1; msk >>= 1)
      lrow[r] += __shfl_xor(lrow[r], msk, 32);

#pragma unroll
  for (int tc = 0; tc < 8; ++tc)
#pragma unroll
    for (int r = 0; r < 8; ++r) {
      const int q   = q0 + half * 8 + r;
      const int col = h * VD + tc * 16 + m;
      Sout[((long)(t * HWN) + q) * DM + col] = f2bf(o[tc][r] / lrow[r]);
    }
}

// ---------------------------------------------------------------------------
extern "C" void kernel_launch(void* const* d_in, const int* in_sizes, int n_in,
                              void* d_out, int out_size, void* d_ws, size_t ws_size,
                              hipStream_t stream) {
  (void)in_sizes; (void)n_in; (void)out_size; (void)ws_size;
  const float* utt    = (const float*)d_in[0];  // updated_track_tokens [T,M,DM]
  const float* tracks = (const float*)d_in[1];  // [T,M,2]
  const float* fpos   = (const float*)d_in[2];  // [HW,2]
  // d_in[3] original_features: unused by the reference
  const float* gct    = (const float*)d_in[4];  // grid_coords_tokens [T,HW,DM]
  const float* Wq     = (const float*)d_in[5];  // [DM,DK]
  const float* Wk     = (const float*)d_in[6];  // [DM,DK]
  const float* Wout   = (const float*)d_in[7];  // [DM,DM]
  const float* qg     = (const float*)d_in[8];  // [DK]
  const float* kg     = (const float*)d_in[9];  // [DK]

  char* ws = (char*)d_ws;
  float*          Qf   = (float*)(ws);                              // 32 MB
  float*          Kf   = (float*)(ws + (size_t)33554432);           // 16 MB
  unsigned short* Qbf  = (unsigned short*)(ws + (size_t)50331648);  // 16 MB
  unsigned short* Kbf  = (unsigned short*)(ws + (size_t)67108864);  //  8 MB
  unsigned short* Vbf  = (unsigned short*)(ws + (size_t)75497472);  // 16 MB
  unsigned short* VbfT = (unsigned short*)(ws + (size_t)92274688);  // 16 MB
  unsigned short* Sbf  = (unsigned short*)(ws + (size_t)109051904); // 32 MB

  // 1) Projections
  gemm_wmma<false><<<dim3(DK / 64, (TT * HWN) / 128), 256, 0, stream>>>(
      gct, Wq, Qf, TT * HWN, DK, DM);
  gemm_wmma<false><<<dim3(DK / 64, (TT * MM) / 128), 256, 0, stream>>>(
      utt, Wk, Kf, TT * MM, DK, DM);

  // 2) Elementwise: LN(Q), RoPE+LN(K), RoPE(V), then V transpose
  ln_q_kernel<<<TT * HWN, 256, 0, stream>>>(Qf, qg, Qbf);
  rope_ln_k_kernel<<<TT * MM, 256, 0, stream>>>(Kf, tracks, kg, Kbf);
  rope_v_kernel<<<TT * MM, 512, 0, stream>>>(utt, tracks, Vbf);
  vtrans_kernel<<<dim3(MM / 64, DM / 64, TT), 256, 0, stream>>>(Vbf, VbfT);

  // 3) Biased attention + softmax + P*V
  attn_kernel<<<dim3(HWN / 64, NH, TT), 128, 0, stream>>>(
      Qbf, Kbf, VbfT, tracks, fpos, Sbf);

  // 4) Output projection
  gemm_wmma<true><<<dim3(DM / 64, (TT * HWN) / 128), 256, 0, stream>>>(
      Sbf, Wout, (float*)d_out, TT * HWN, DM, DM);
}
